// Separator_64896955842665
// MI455X (gfx1250) — compile-verified, matrix operand order checked by reading, not run on hardware
//
#include <hip/hip_runtime.h>
#include <hip/hip_bf16.h>

typedef __attribute__((ext_vector_type(16))) _Float16 v16h;
typedef __attribute__((ext_vector_type(8)))  _Float16 v8h;
typedef __attribute__((ext_vector_type(8)))  float    v8f;
typedef unsigned int v4u __attribute__((ext_vector_type(4)));
typedef unsigned int v8u __attribute__((ext_vector_type(8)));

#define B_     16
#define T_     64000
#define T1     32000   // conv1 out length
#define T2     16000   // conv2 out length
#define T3     8000    // conv3 out length
#define T1P    (T1 + 16)  // padded rows (8 front, 8 back)
#define T2P    (T2 + 16)
#define C1     32
#define C2     64
#define C3     128

// ---------------- workspace layout (bytes) ----------------
#define OFF_X1     0
#define SZ_X1      (B_*(size_t)T1P*C1*2)
#define OFF_X2     (OFF_X1 + SZ_X1)
#define SZ_X2      (B_*(size_t)T2P*C2*2)
#define OFF_POOL   (OFF_X2 + SZ_X2)
#define SZ_POOL    (B_*C3*4)
#define OFF_FAMPS  (OFF_POOL + SZ_POOL)
#define SZ_FAMPS   (B_*128*4)
#define OFF_W2P    (OFF_FAMPS + SZ_FAMPS)
#define SZ_W2P     (9*64*32*2)
#define OFF_W3P    (OFF_W2P + SZ_W2P)

// ---- CDNA5 async global->LDS copy (ASYNCcnt tracked) ----
__device__ inline void async_g2l_b128(unsigned int lds_off, const void* gaddr) {
  asm volatile("global_load_async_to_lds_b128 %0, %1, off"
               :: "v"(lds_off), "v"(gaddr) : "memory");
}
__device__ inline void wait_async0() {
  asm volatile("s_wait_asynccnt 0" ::: "memory");
}

// ---- CDNA5 Tensor Data Mover: one-wave 2D strided DMA to LDS ----
// D# per CDNA5 ISA; groups 2/3 zero (<=2D tensor).
// TENSORcnt tracked; EXEC ignored.
__device__ inline void tdm_load_to_lds(v4u g0, v8u g1, v4u g2, v4u g3) {
  asm volatile("tensor_load_to_lds %0, %1, %2, %3"
               :: "s"(g0), "s"(g1), "s"(g2), "s"(g3) : "memory");
}

// Zero x1 + x2 + pooled (pads must be 0; pooled is an atomic accumulator).
__global__ void zero_ws_k(uint4* __restrict__ p, long n16) {
  long i = blockIdx.x * (long)blockDim.x + threadIdx.x;
  long stride = (long)gridDim.x * blockDim.x;
  uint4 z; z.x = 0u; z.y = 0u; z.z = 0u; z.w = 0u;
  for (; i < n16; i += stride) p[i] = z;
}

// Repack conv weights f32 (O,I,K) -> f16 [k][co][ci].
__global__ void prep_w_k(const float* __restrict__ w2, const float* __restrict__ w3,
                         _Float16* __restrict__ w2p, _Float16* __restrict__ w3p) {
  int i = blockIdx.x * blockDim.x + threadIdx.x;
  if (i < 9*64*32) {
    int k = i / (64*32); int r = i % (64*32); int co = r / 32; int ci = r % 32;
    w2p[i] = (_Float16)w2[co*(32*9) + ci*9 + k];
  } else if (i < 9*64*32 + 9*128*64) {
    int j = i - 9*64*32;
    int k = j / (128*64); int r = j % (128*64); int co = r / 64; int ci = r % 64;
    w3p[j] = (_Float16)w3[co*(64*9) + ci*9 + k];
  }
}

// conv1: 1 -> 32 channels, direct VALU. One thread per (b, t_out).
__global__ void conv1_k(const float* __restrict__ audio, const float* __restrict__ w1,
                        const float* __restrict__ b1, _Float16* __restrict__ x1) {
  __shared__ float sW[C1*9];
  __shared__ float sB[C1];
  for (int i = threadIdx.x; i < C1*9; i += blockDim.x) sW[i] = w1[i];
  for (int i = threadIdx.x; i < C1;   i += blockDim.x) sB[i] = b1[i];
  __syncthreads();
  int g = blockIdx.x * blockDim.x + threadIdx.x;   // 16*32000 threads
  int b = g / T1, t = g % T1;
  float s[9];
  int base = 2*t - 4;
  #pragma unroll
  for (int k = 0; k < 9; ++k) {
    int idx = base + k;
    s[k] = (idx >= 0 && idx < T_) ? audio[(size_t)b*T_ + idx] : 0.0f;
  }
  _Float16* o = x1 + ((size_t)b*T1P + (t + 8)) * C1;
  #pragma unroll
  for (int chunk = 0; chunk < 4; ++chunk) {
    v8h ov;
    #pragma unroll
    for (int j = 0; j < 8; ++j) {
      int co = chunk*8 + j;
      float a = sB[co];
      #pragma unroll
      for (int k = 0; k < 9; ++k) a = fmaf(sW[co*9 + k], s[k], a);
      a = a > 0.0f ? a : 0.0f;
      ov[j] = (_Float16)a;
    }
    *(v8h*)(o + chunk*8) = ov;
  }
}

// Build a 16x32 f16 WMMA fragment from two b128 loads (global or LDS).
__device__ inline v16h load_frag(const _Float16* p) {
  v8h lo = *(const v8h*)p;
  v8h hi = *(const v8h*)(p + 16);
  v16h f;
  #pragma unroll
  for (int i = 0; i < 8; ++i) { f[i] = lo[i]; f[i + 8] = hi[i]; }
  return f;
}

__device__ inline v8f wmma_f16(v16h a, v16h b, v8f c) {
  return __builtin_amdgcn_wmma_f32_16x16x32_f16(false, a, false, b, (short)0, c,
                                                false, false);
}

// conv2: implicit GEMM. Wave: M=64 (4 tiles) x N=32 (2 tiles), K=9 taps x 32 ci.
// Weights async-staged to LDS; k-loop fully unrolled so all loads use a single
// base pointer + immediate offsets (no per-tap 64-bit address VALU).
__global__ void __launch_bounds__(256)
conv2_wmma_k(const _Float16* __restrict__ x1, const _Float16* __restrict__ w2p,
             const float* __restrict__ b2, _Float16* __restrict__ x2) {
  __shared__ _Float16 sW[9*64*32];                      // 36,864 B
  for (int i = threadIdx.x; i < (9*64*32)/8; i += 256) {
    unsigned int dst = (unsigned int)(uintptr_t)(&sW[0]) + i*16u;
    async_g2l_b128(dst, w2p + i*8);
  }
  wait_async0();
  __syncthreads();

  const int wave = blockIdx.x * 8 + (threadIdx.x >> 5); // 8000 waves
  const int lane = threadIdx.x & 31;
  const int l16 = lane & 15, hi = lane >> 4;
  const int b  = wave / (T2/32);                        // T2/32 = 500
  const int t0 = (wave % (T2/32)) * 32;

  // Per-lane, loop-invariant bases; everything below is immediate offsets.
  const _Float16* bp = x1 + (size_t)b*T1P*C1 + (size_t)(2*(t0 + l16) + 4)*C1 + hi*8;
  const _Float16* ap = &sW[l16*32 + hi*8];
  __builtin_prefetch(bp, 0, 0);
  __builtin_prefetch(bp + 32*C1, 0, 0);

  v8f acc[2][4] = {};
  #pragma unroll
  for (int k = 0; k < 9; ++k) {
    v16h bfrag[2];
    #pragma unroll
    for (int n = 0; n < 2; ++n)                         // n*32 rows, k rows
      bfrag[n] = load_frag(bp + (n*32 + k)*C1);
    #pragma unroll
    for (int m = 0; m < 4; ++m) {
      v16h afrag = load_frag(ap + (k*64 + m*16)*32);
      #pragma unroll
      for (int n = 0; n < 2; ++n) acc[n][m] = wmma_f16(afrag, bfrag[n], acc[n][m]);
    }
  }
  #pragma unroll
  for (int n = 0; n < 2; ++n) {
    const int tout = t0 + n*16 + l16;
    _Float16* xo = x2 + ((size_t)b*T2P + (tout + 8)) * C2;
    #pragma unroll
    for (int m = 0; m < 4; ++m) {
      int cobase = m*16 + hi*8;
      v8h ov;
      #pragma unroll
      for (int j = 0; j < 8; ++j) {
        float v = acc[n][m][j] + b2[cobase + j];
        v = v > 0.0f ? v : 0.0f;
        ov[j] = (_Float16)v;
      }
      *(v8h*)(xo + cobase) = ov;
    }
  }
}

// conv3: implicit GEMM fused with mean-pool. Block-uniform mhalf selects
// co 0..63 or 64..127; that half of w3p (72 KB, 9 rows x 8 KB with 16 KB pitch)
// is staged to LDS by a single TDM descriptor issued by wave 0.
__global__ void __launch_bounds__(256)
conv3_wmma_k(const _Float16* __restrict__ x2, const _Float16* __restrict__ w3p,
             const float* __restrict__ b3, float* __restrict__ pooled) {
  __shared__ _Float16 sW[9*64*64];                      // 73,728 B
  const int mhalf = blockIdx.x & 1;
  const int bb    = blockIdx.x >> 1;                    // 0..499

  if ((threadIdx.x >> 5) == 0) {
    // D# group0: count=1 | lds_addr | global_addr(57b) | type=2
    unsigned int lds0 = __builtin_amdgcn_readfirstlane(
        (unsigned int)(uintptr_t)(&sW[0]));
    unsigned long long ga = (unsigned long long)(uintptr_t)(w3p + mhalf*4096);
    v4u g0; g0[0] = 1u; g0[1] = lds0;
    g0[2] = (unsigned int)ga;
    g0[3] = ((unsigned int)(ga >> 32) & 0x01FFFFFFu) | (2u << 30);
    // D# group1: data_size=8B(code 3); tensor_dim0=1024, tensor_dim1=9;
    // tile_dim0=1024, tile_dim1=9; tensor_dim0_stride=2048 (16 KB pitch).
    v8u g1;
    g1[0] = 3u << 16;                                   // data_size
    g1[1] = (1024u & 0xFFFFu) << 16;                    // tensor_dim0 lo16
    g1[2] = (1024u >> 16) | (9u << 16);                 // td0 hi16 | td1 lo16
    g1[3] = (9u >> 16) | (1024u << 16);                 // td1 hi16 | tile_dim0
    g1[4] = 9u;                                         // tile_dim1 | tile_dim2=0
    g1[5] = 2048u;                                      // td0_stride lo32
    g1[6] = 0u;                                         // td0_stride hi16 | td1_stride lo16
    g1[7] = 0u;
    v4u gz; gz[0] = 0u; gz[1] = 0u; gz[2] = 0u; gz[3] = 0u;
    tdm_load_to_lds(g0, g1, gz, gz);
    __builtin_amdgcn_s_wait_tensorcnt(0);
  }
  __syncthreads();

  const int id   = bb*8 + (threadIdx.x >> 5);           // 0..3999
  const int lane = threadIdx.x & 31;
  const int l16 = lane & 15, hi = lane >> 4;
  const int b  = id / (T3/32);                          // T3/32 = 250
  const int t0 = (id % (T3/32)) * 32;

  const _Float16* bp = x2 + (size_t)b*T2P*C2 + (size_t)(2*(t0 + l16) + 4)*C2 + hi*8;
  const _Float16* ap = &sW[l16*64 + hi*8];
  __builtin_prefetch(bp, 0, 0);
  __builtin_prefetch(bp + 32*C2, 0, 0);

  v8f acc[2][4] = {};
  #pragma unroll
  for (int k = 0; k < 9; ++k) {
    #pragma unroll
    for (int kb = 0; kb < 2; ++kb) {
      v16h bfrag[2];
      #pragma unroll
      for (int n = 0; n < 2; ++n)
        bfrag[n] = load_frag(bp + (n*32 + k)*C2 + kb*32);
      #pragma unroll
      for (int m = 0; m < 4; ++m) {
        v16h afrag = load_frag(ap + (k*64 + m*16)*64 + kb*32);
        #pragma unroll
        for (int n = 0; n < 2; ++n) acc[n][m] = wmma_f16(afrag, bfrag[n], acc[n][m]);
      }
    }
  }
  // Epilogue: bias + relu per t element, sum the two N-tiles, reduce over the
  // 16-lane t-group, single atomicAdd per (co) strip.
  #pragma unroll
  for (int m = 0; m < 4; ++m) {
    int cobase = mhalf*64 + m*16 + hi*8;
    #pragma unroll
    for (int j = 0; j < 8; ++j) {
      float bias = b3[cobase + j];
      float v0 = acc[0][m][j] + bias; v0 = v0 > 0.0f ? v0 : 0.0f;
      float v1 = acc[1][m][j] + bias; v1 = v1 > 0.0f ? v1 : 0.0f;
      float v = v0 + v1;
      v += __shfl_xor(v, 1, 32);
      v += __shfl_xor(v, 2, 32);
      v += __shfl_xor(v, 4, 32);
      v += __shfl_xor(v, 8, 32);                        // sum over 16 t's
      if (l16 == 0) atomicAdd(&pooled[b*C3 + cobase + j], v);
    }
  }
}

// Linear head + amp normalization. Block per batch, 128 threads.
__global__ void feats_k(const float* __restrict__ pooled, const float* __restrict__ wl,
                        const float* __restrict__ bl, float* __restrict__ famps) {
  __shared__ float sP[128], sF[128], sSum;
  const int b = blockIdx.x, j = threadIdx.x;
  sP[j] = pooled[b*128 + j] * (1.0f / (float)T3);       // mean over 8000
  __syncthreads();
  float acc = bl[j];
  #pragma unroll 8
  for (int c = 0; c < 128; ++c) acc = fmaf(sP[c], wl[j*128 + c], acc);
  sF[j] = acc;
  __syncthreads();
  if (j == 0) {
    float s = 0.0f;
    for (int h = 0; h < 64; ++h) { float a = sF[h] > 0.0f ? sF[h] : 0.0f; s += a; }
    sSum = s;
  }
  __syncthreads();
  if (j < 64) {
    float a = sF[j] > 0.0f ? sF[j] : 0.0f;
    famps[b*128 + j] = a / (sSum + 1e-6f);
  } else {
    famps[b*128 + j] = sF[j];                           // raw noise magnitudes
  }
}

// Harmonic bank via Chebyshev recurrence + shaped noise. Thread per (b,t).
__global__ void synth_k(const float* __restrict__ f0, const float* __restrict__ wn,
                        const float* __restrict__ famps, float* __restrict__ out) {
  __shared__ float sA[64], sN[64];
  const int b = blockIdx.x / (T_/256);
  const int t = (blockIdx.x % (T_/256)) * 256 + threadIdx.x;
  if (threadIdx.x < 64)        sA[threadIdx.x]      = famps[b*128 + threadIdx.x];
  else if (threadIdx.x < 128)  sN[threadIdx.x - 64] = famps[b*128 + threadIdx.x];
  __syncthreads();
  // jnp.linspace(0, T/SR, T) includes the endpoint: dt = (T/SR)/(T-1)
  const float dt = 4.0f / 63999.0f;
  const float f  = __builtin_nontemporal_load(&f0[(size_t)b*T_ + t]);
  const float w  = __builtin_nontemporal_load(&wn[(size_t)b*T_ + t]);
  const float theta = 6.28318530717958647f * f * ((float)t * dt);
  float s, c;
  __sincosf(theta, &s, &c);
  const float c2 = 2.0f * c;
  float sPrev = 0.0f, sCur = s, sum = 0.0f;             // sin(0), sin(theta)
  #pragma unroll 8
  for (int h = 0; h < 64; ++h) {                        // harmonic h+1
    sum = fmaf(sA[h], sCur, sum);
    float nx = fmaf(c2, sCur, -sPrev);                  // sin((h+2)theta)
    sPrev = sCur; sCur = nx;
  }
  const int bin = t / (T_ / 64);                        // bin_size = 1000
  __builtin_nontemporal_store(sum + sN[bin] * w, &out[(size_t)b*T_ + t]);
}

extern "C" void kernel_launch(void* const* d_in, const int* in_sizes, int n_in,
                              void* d_out, int out_size, void* d_ws, size_t ws_size,
                              hipStream_t stream) {
  (void)in_sizes; (void)n_in; (void)out_size; (void)ws_size;
  const float* audio = (const float*)d_in[0];
  const float* f0    = (const float*)d_in[1];
  const float* wn    = (const float*)d_in[2];
  const float* w1    = (const float*)d_in[3];
  const float* b1    = (const float*)d_in[4];
  const float* w2    = (const float*)d_in[5];
  const float* b2    = (const float*)d_in[6];
  const float* w3    = (const float*)d_in[7];
  const float* b3    = (const float*)d_in[8];
  const float* wl    = (const float*)d_in[9];
  const float* bl    = (const float*)d_in[10];
  float* out = (float*)d_out;

  char* ws = (char*)d_ws;
  _Float16* x1     = (_Float16*)(ws + OFF_X1);
  _Float16* x2     = (_Float16*)(ws + OFF_X2);
  float*    pooled = (float*)   (ws + OFF_POOL);
  float*    famps  = (float*)   (ws + OFF_FAMPS);
  _Float16* w2p    = (_Float16*)(ws + OFF_W2P);
  _Float16* w3p    = (_Float16*)(ws + OFF_W3P);

  const long n16 = (long)(SZ_X1 + SZ_X2 + SZ_POOL) / 16;
  zero_ws_k<<<4096, 256, 0, stream>>>((uint4*)ws, n16);
  prep_w_k<<<(9*64*32 + 9*128*64 + 255)/256, 256, 0, stream>>>(w2, w3, w2p, w3p);
  conv1_k<<<(B_*T1)/256, 256, 0, stream>>>(audio, w1, b1, x1);
  conv2_wmma_k<<<(B_*(T2/32))/8, 256, 0, stream>>>(x1, w2p, b2, x2);        // 1000 blocks
  conv3_wmma_k<<<(B_*(T3/32)*2)/8, 256, 0, stream>>>(x2, w3p, b3, pooled);  // 1000 blocks
  feats_k<<<B_, 128, 0, stream>>>(pooled, wl, bl, famps);
  synth_k<<<B_*(T_/256), 256, 0, stream>>>(f0, wn, famps, out);
}
